// CoarseGraphDecoder_86225763435148
// MI455X (gfx1250) — compile-verified, matrix operand order checked by reading, not run on hardware
//
#include <hip/hip_runtime.h>
#include <hip/hip_bf16.h>
#include <stdint.h>

// ---------------- types for WMMA ----------------
typedef __attribute__((ext_vector_type(16))) __bf16       v16bf;
typedef __attribute__((ext_vector_type(4)))  __bf16       v4bf;
typedef __attribute__((ext_vector_type(8)))  float        v8f;
typedef __attribute__((ext_vector_type(4)))  unsigned int v4u;

union FragU { struct { v4u lo, hi; } p; v16bf v; };

#define NCc   842
#define NFc   16384
#define EPSc  1e-5f

// LDS pitches (elements). +32B skew per row spreads banks for b128 loads.
#define P_BF  272   // bf16 pitch for 256-wide activations (544 B rows)
#define P_F   264   // f32 pitch (1056 B rows)
#define P_NI  528   // bf16 pitch for 512-wide node input (1056 B rows)

// ---- fused-kernel dynamic LDS layout (bytes) ----
#define O_H1     0                       // 112*272*2 = 60928 bf16
#define O_H2     60928                   // 112*272*2 = 60928 bf16
#define O_H3     121856                  // 112*264*4 = 118272 f32
// node-phase buffers overlay dead H1/H2 region:
#define O_AGG    0                       // 16*256*4  = 16384 f32
#define O_NIN    16384                   // 16*528*2  = 16896 bf16
#define O_HN1    33280                   // 16*272*2  = 8704 bf16
#define O_HN2    41984                   // 16*272*2  = 8704 bf16
#define O_HN3    50688                   // 16*264*4  = 16896 f32 (into dead H2)
#define O_ESRC   240128                  // 112*4
#define O_EA0    240576                  // 112*4
#define O_EA1    241024                  // 112*4
#define O_WC0    241472                  // 256*4
#define O_WC1    242496                  // 256*4
#define SMEM_FUSED 243520

// Load a 16x32 bf16 A-fragment (or 32x16 B-fragment from N-major-transposed
// weights) per the CDNA5 16-bit A/B VGPR layout:
//   lane 0-15: row m=lane,    K = {0..7} then {16..23}
//   lane16-31: row m=lane-16, K = {8..15} then {24..31}
// -> two contiguous 16-byte chunks per lane.
__device__ __forceinline__ v16bf ldfrag(const __bf16* base, int pitch, int row0,
                                        int k0, int lane) {
  const __bf16* p = base + (size_t)(row0 + (lane & 15)) * pitch
                         + (k0 + ((lane >> 4) << 3));
  FragU f;
  f.p.lo = *(const v4u*)p;
  f.p.hi = *(const v4u*)(p + 16);
  return f.v;
}

__device__ __forceinline__ v8f wmma_bf16(v16bf a, v16bf b, v8f c) {
  return __builtin_amdgcn_wmma_f32_16x16x32_bf16(
      /*neg_a=*/false, a, /*neg_b=*/false, b,
      /*c_mod=*/(short)0, c, /*reuse_a=*/false, /*reuse_b=*/false);
}

__device__ __forceinline__ v4bf cvt4(float4 v) {
  v4bf r;
  r[0] = (__bf16)v.x; r[1] = (__bf16)v.y; r[2] = (__bf16)v.z; r[3] = (__bf16)v.w;
  return r;
}

// ---------------- weight transpose + fp32 -> bf16 ----------------
// src: K x N row-major fp32 ; dst: N x K row-major bf16
__global__ __launch_bounds__(256) void transpose_bf16(const float* __restrict__ src,
                                                      __bf16* __restrict__ dst,
                                                      int K, int N) {
  int id = blockIdx.x * 256 + threadIdx.x;
  if (id < K * N) {
    int n = id / K, k = id - n * K;
    dst[id] = (__bf16)src[(size_t)k * N + n];
  }
}

// ---------------- precompute GEMM: out[M x 256] = A[M x 256] @ W + bias ----
// A fp32 row-major (staged to bf16 LDS), Wt bf16 N-major (256 x 256).
// Output rows are padded to a multiple of 16: stores are unconditional.
__global__ __launch_bounds__(256) void wmma_gemm_rowtile(
    const float* __restrict__ A, const __bf16* __restrict__ Wt,
    const float* __restrict__ bias, float* __restrict__ out, int M) {
  __shared__ __bf16 As[16 * P_BF];
  int m0 = blockIdx.x * 16;
  for (int idx = threadIdx.x; idx < 16 * 64; idx += 256) {   // float4 chunks
    int r = idx >> 6, c4 = idx & 63;
    float4 v = make_float4(0.f, 0.f, 0.f, 0.f);
    if (m0 + r < M) v = ((const float4*)(A + (size_t)(m0 + r) * 256))[c4];
    *(v4bf*)&As[r * P_BF + c4 * 4] = cvt4(v);
  }
  __syncthreads();
  int wave = threadIdx.x >> 5, lane = threadIdx.x & 31;
  int col16 = lane & 15, rsel = (lane >> 4) << 3;
  for (int nt = wave * 2; nt < wave * 2 + 2; ++nt) {
    int n0 = nt << 4, col = n0 + col16;
    float bv = bias ? bias[col] : 0.f;
    v8f acc;
#pragma unroll
    for (int v = 0; v < 8; ++v) acc[v] = bv;
#pragma unroll
    for (int k = 0; k < 8; ++k) {
      v16bf Af = ldfrag(As, P_BF, 0, k * 32, lane);
      v16bf Bf = ldfrag(Wt, 256, n0, k * 32, lane);
      acc = wmma_bf16(Af, Bf, acc);
    }
    int mb = m0 + rsel;
#pragma unroll
    for (int v = 0; v < 8; ++v) out[(size_t)(mb + v) * 256 + col] = acc[v];
  }
}

// ---------------- fused kernel helpers ----------------
// 112x256 = 112x256 @ 256x256 (+bias, opt relu). 8 waves x 2 N-tiles,
// B-fragments register-cached across the 7 M-tiles.
__device__ __forceinline__ void edge_gemm(const __bf16* Hin, const __bf16* Wt,
                                          const float* bias, __bf16* outBf,
                                          float* outF, bool relu, int tid) {
  int wave = tid >> 5, lane = tid & 31;
  int col16 = lane & 15, rsel = (lane >> 4) << 3;
  for (int nt = wave * 2; nt < wave * 2 + 2; ++nt) {
    int n0 = nt << 4, col = n0 + col16;
    v16bf Bf[8];
#pragma unroll
    for (int k = 0; k < 8; ++k) Bf[k] = ldfrag(Wt, 256, n0, k * 32, lane);
    float bv = bias[col];
    for (int mt = 0; mt < 7; ++mt) {
      v8f acc;
#pragma unroll
      for (int v = 0; v < 8; ++v) acc[v] = bv;
#pragma unroll
      for (int k = 0; k < 8; ++k) {
        v16bf Af = ldfrag(Hin, P_BF, mt * 16, k * 32, lane);
        acc = wmma_bf16(Af, Bf[k], acc);
      }
      int mb = mt * 16 + rsel;
      if (outBf) {
#pragma unroll
        for (int v = 0; v < 8; ++v) {
          float x = acc[v];
          if (relu) x = fmaxf(x, 0.f);
          outBf[(mb + v) * P_BF + col] = (__bf16)x;
        }
      } else {
#pragma unroll
        for (int v = 0; v < 8; ++v) outF[(mb + v) * P_F + col] = acc[v];
      }
    }
  }
}

// 16xK input, 16x256 output.
__device__ __forceinline__ void node_gemm(const __bf16* In, int pitchIn, int K,
                                          const __bf16* Wt, const float* bias,
                                          __bf16* outBf, float* outF, bool relu,
                                          int tid) {
  int wave = tid >> 5, lane = tid & 31;
  int col16 = lane & 15, rsel = (lane >> 4) << 3;
  for (int nt = wave * 2; nt < wave * 2 + 2; ++nt) {
    int n0 = nt << 4, col = n0 + col16;
    float bv = bias[col];
    v8f acc;
#pragma unroll
    for (int v = 0; v < 8; ++v) acc[v] = bv;
    for (int k = 0; k < K; k += 32) {
      v16bf Af = ldfrag(In, pitchIn, 0, k, lane);
      v16bf Bf = ldfrag(Wt, K, n0, k, lane);
      acc = wmma_bf16(Af, Bf, acc);
    }
    if (outBf) {
#pragma unroll
      for (int v = 0; v < 8; ++v) {
        float x = acc[v];
        if (relu) x = fmaxf(x, 0.f);
        outBf[(rsel + v) * P_BF + col] = (__bf16)x;
      }
    } else {
#pragma unroll
      for (int v = 0; v < 8; ++v) outF[(rsel + v) * P_F + col] = acc[v];
    }
  }
}

// Wave-parallel LayerNorm: each wave takes every-8th row; each lane owns 8
// contiguous elements (2x float4), 5-step shfl_xor butterfly for mean/var.
__device__ __forceinline__ void layernorm_rows_wave(float* H, int rows,
                                                    const float* g,
                                                    const float* be, int tid) {
  int wave = tid >> 5, lane = tid & 31;
  const float4* gv  = (const float4*)g;
  const float4* bev = (const float4*)be;
  float4 g0  = gv[lane * 2],  g1  = gv[lane * 2 + 1];
  float4 be0 = bev[lane * 2], be1 = bev[lane * 2 + 1];
  for (int row = wave; row < rows; row += 8) {
    float4* r = (float4*)(H + row * P_F);
    float4 a = r[lane * 2], b = r[lane * 2 + 1];
    float s  = a.x + a.y + a.z + a.w + b.x + b.y + b.z + b.w;
    float ss = a.x * a.x + a.y * a.y + a.z * a.z + a.w * a.w +
               b.x * b.x + b.y * b.y + b.z * b.z + b.w * b.w;
#pragma unroll
    for (int m = 16; m >= 1; m >>= 1) {
      s  += __shfl_xor(s, m, 32);
      ss += __shfl_xor(ss, m, 32);
    }
    float mu  = s * (1.f / 256.f);
    float var = ss * (1.f / 256.f) - mu * mu;
    float rs  = rsqrtf(var + EPSc);
    a.x = (a.x - mu) * rs * g0.x + be0.x;  a.y = (a.y - mu) * rs * g0.y + be0.y;
    a.z = (a.z - mu) * rs * g0.z + be0.z;  a.w = (a.w - mu) * rs * g0.w + be0.w;
    b.x = (b.x - mu) * rs * g1.x + be1.x;  b.y = (b.y - mu) * rs * g1.y + be1.y;
    b.z = (b.z - mu) * rs * g1.z + be1.z;  b.w = (b.w - mu) * rs * g1.w + be1.w;
    r[lane * 2] = a; r[lane * 2 + 1] = b;
  }
}

// ---------------- fused edge-MLP + aggregate + node-MLP ----------------
// grid = B * (NF/16); block = 256 threads (8 waves). Workgroup handles
// 16 fine nodes = 112 consecutive edges (edge_dst = repeat(arange(NF),7)).
__global__ __launch_bounds__(256) void fused_edge_node(
    const float* __restrict__ srcPre, const float* __restrict__ dstPre,
    const __bf16* __restrict__ W2t, const __bf16* __restrict__ W3t,
    const __bf16* __restrict__ W1nt, const __bf16* __restrict__ W2nt,
    const __bf16* __restrict__ W3nt,
    const float* __restrict__ eb1, const float* __restrict__ eb2,
    const float* __restrict__ eg, const float* __restrict__ ebe,
    const float* __restrict__ nb0, const float* __restrict__ nb1,
    const float* __restrict__ nb2,
    const float* __restrict__ ng, const float* __restrict__ nbe,
    const float* __restrict__ latlon, const int* __restrict__ edge_src,
    const float* __restrict__ edge_attr, const float* __restrict__ eW0,
    float* __restrict__ out) {
  extern __shared__ char smem[];
  __bf16* H1     = (__bf16*)(smem + O_H1);
  __bf16* H2     = (__bf16*)(smem + O_H2);
  float*  H3     = (float*)(smem + O_H3);
  float*  agg    = (float*)(smem + O_AGG);
  __bf16* NodeIn = (__bf16*)(smem + O_NIN);
  __bf16* Hn1    = (__bf16*)(smem + O_HN1);
  __bf16* Hn2    = (__bf16*)(smem + O_HN2);
  float*  Hn3    = (float*)(smem + O_HN3);
  int*    esrc_s = (int*)(smem + O_ESRC);
  float*  ea0_s  = (float*)(smem + O_EA0);
  float*  ea1_s  = (float*)(smem + O_EA1);
  float*  wc0    = (float*)(smem + O_WC0);
  float*  wc1    = (float*)(smem + O_WC1);

  int tid   = threadIdx.x;
  int bidx  = blockIdx.x >> 10;         // batch (0..1); 1024 groups per batch
  int group = blockIdx.x & 1023;
  int node0 = group << 4;
  int e0    = group * 112;

  // stage edge metadata + W1c rows
  if (tid < 112) {
    esrc_s[tid] = edge_src[e0 + tid];
    ea0_s[tid]  = edge_attr[(size_t)(e0 + tid) * 2 + 0];
    ea1_s[tid]  = edge_attr[(size_t)(e0 + tid) * 2 + 1];
  }
  wc0[tid] = eW0[512 * 256 + tid];
  wc1[tid] = eW0[513 * 256 + tid];
  __syncthreads();

  // ---- edge layer 0 (decomposed): H1 = relu(srcPre[s] + dstPre[d] + ea@W1c)
  // float4-vectorized: thread = (node-group g4, float4-chunk c of 64).
  {
    const float* sp = srcPre + (size_t)bidx * NCc * 256;
    int c  = tid & 63;        // float4 chunk: features 4c..4c+3
    int g4 = tid >> 6;        // node phase 0..3
    float4 w0 = ((const float4*)wc0)[c];
    float4 w1 = ((const float4*)wc1)[c];
    for (int j = g4; j < 16; j += 4) {
      float4 dv = ((const float4*)(dstPre + (size_t)(node0 + j) * 256))[c];
      int i = j * 7;
      for (int t7 = 0; t7 < 7; ++t7, ++i) {
        float4 sv = ((const float4*)(sp + (size_t)esrc_s[i] * 256))[c];
        float e0a = ea0_s[i], e1a = ea1_s[i];
        float4 v;
        v.x = fmaxf(sv.x + dv.x + e0a * w0.x + e1a * w1.x, 0.f);
        v.y = fmaxf(sv.y + dv.y + e0a * w0.y + e1a * w1.y, 0.f);
        v.z = fmaxf(sv.z + dv.z + e0a * w0.z + e1a * w1.z, 0.f);
        v.w = fmaxf(sv.w + dv.w + e0a * w0.w + e1a * w1.w, 0.f);
        *(v4bf*)&H1[i * P_BF + c * 4] = cvt4(v);
      }
    }
  }
  __syncthreads();
  edge_gemm(H1, W2t, eb1, H2, nullptr, true, tid);   // layer 1 + relu
  __syncthreads();
  edge_gemm(H2, W3t, eb2, nullptr, H3, false, tid);  // layer 2 (pre-LN, fp32)
  __syncthreads();
  layernorm_rows_wave(H3, 112, eg, ebe, tid);
  __syncthreads();

  // ---- aggregate: sum 7 consecutive LN'd edge rows per node (float4)
  {
    int j = tid >> 4, c0 = (tid & 15) * 4;   // 4 float4 chunks per thread
#pragma unroll
    for (int cc = 0; cc < 4; ++cc) {
      float4 s = make_float4(0.f, 0.f, 0.f, 0.f);
      for (int r = 0; r < 7; ++r) {
        float4 v = ((const float4*)(H3 + (j * 7 + r) * P_F))[c0 + cc];
        s.x += v.x; s.y += v.y; s.z += v.z; s.w += v.w;
      }
      ((float4*)(agg + j * 256))[c0 + cc] = s;
    }
  }
  __syncthreads();

  // ---- node input = [latlon(dst) | agg]  (16 x 512 bf16, float4 chunks)
  for (int idx = tid; idx < 16 * 128; idx += 256) {
    int row = idx >> 7, c4 = idx & 127, col = c4 * 4;
    float4 v = (col < 256)
        ? ((const float4*)(latlon + (size_t)(node0 + row) * 256))[c4]
        : ((const float4*)(agg + row * 256))[c4 - 64];
    *(v4bf*)&NodeIn[row * P_NI + col] = cvt4(v);
  }
  __syncthreads();
  node_gemm(NodeIn, P_NI, 512, W1nt, nb0, Hn1, nullptr, true, tid);
  __syncthreads();
  node_gemm(Hn1, P_BF, 256, W2nt, nb1, Hn2, nullptr, true, tid);
  __syncthreads();
  node_gemm(Hn2, P_BF, 256, W3nt, nb2, nullptr, Hn3, false, tid);
  __syncthreads();
  layernorm_rows_wave(Hn3, 16, ng, nbe, tid);
  __syncthreads();

  // ---- write 16 x 256 output rows (float4, coalesced)
  {
    float4* ob = (float4*)(out + ((size_t)bidx * NFc + node0) * 256);
    for (int idx = tid; idx < 16 * 64; idx += 256) {
      int row = idx >> 6, c = idx & 63;
      ob[idx] = ((const float4*)(Hn3 + row * P_F))[c];
    }
  }
}

// ---------------- host ----------------
extern "C" void kernel_launch(void* const* d_in, const int* in_sizes, int n_in,
                              void* d_out, int out_size, void* d_ws,
                              size_t ws_size, hipStream_t stream) {
  const float* x       = (const float*)d_in[0];   // (1,2,842,256)
  const float* latlon  = (const float*)d_in[1];   // (16384,256)
  const float* edge_at = (const float*)d_in[2];   // (E,2)
  const float* eW0 = (const float*)d_in[3];  const float* eb0 = (const float*)d_in[4];
  const float* eW1 = (const float*)d_in[5];  const float* eb1 = (const float*)d_in[6];
  const float* eW2 = (const float*)d_in[7];  const float* eb2 = (const float*)d_in[8];
  const float* nW0 = (const float*)d_in[9];  const float* nb0 = (const float*)d_in[10];
  const float* nW1 = (const float*)d_in[11]; const float* nb1 = (const float*)d_in[12];
  const float* nW2 = (const float*)d_in[13]; const float* nb2 = (const float*)d_in[14];
  const float* eg  = (const float*)d_in[15]; const float* ebe = (const float*)d_in[16];
  const float* ng  = (const float*)d_in[17]; const float* nbe = (const float*)d_in[18];
  const int*   edge_src = (const int*)d_in[19];
  (void)in_sizes; (void)n_in; (void)out_size; (void)ws_size;

  char* ws = (char*)d_ws;
  size_t off = 0;
  auto alloc = [&](size_t bytes) -> void* {
    void* p = ws + off;
    off += (bytes + 255) & ~(size_t)255;
    return p;
  };
  float*  srcPre = (float*)alloc((size_t)1696 * 256 * 4);   // padded to 106 tiles
  float*  dstPre = (float*)alloc((size_t)16384 * 256 * 4);
  __bf16* W1at   = (__bf16*)alloc((size_t)256 * 256 * 2);
  __bf16* W1bt   = (__bf16*)alloc((size_t)256 * 256 * 2);
  __bf16* W2t    = (__bf16*)alloc((size_t)256 * 256 * 2);
  __bf16* W3t    = (__bf16*)alloc((size_t)256 * 256 * 2);
  __bf16* W1nt   = (__bf16*)alloc((size_t)256 * 512 * 2);
  __bf16* W2nt   = (__bf16*)alloc((size_t)256 * 256 * 2);
  __bf16* W3nt   = (__bf16*)alloc((size_t)256 * 256 * 2);

  // weight transposes -> bf16 N-major
  transpose_bf16<<<256, 256, 0, stream>>>(eW0,             W1at, 256, 256);
  transpose_bf16<<<256, 256, 0, stream>>>(eW0 + 256 * 256, W1bt, 256, 256);
  transpose_bf16<<<256, 256, 0, stream>>>(eW1,             W2t,  256, 256);
  transpose_bf16<<<256, 256, 0, stream>>>(eW2,             W3t,  256, 256);
  transpose_bf16<<<512, 256, 0, stream>>>(nW0,             W1nt, 512, 256);
  transpose_bf16<<<256, 256, 0, stream>>>(nW1,             W2nt, 256, 256);
  transpose_bf16<<<256, 256, 0, stream>>>(nW2,             W3nt, 256, 256);

  // srcPre = x @ W1a + b1 ; dstPre = latlon @ W1b
  wmma_gemm_rowtile<<<106, 256, 0, stream>>>(x, W1at, eb0, srcPre, 1684);
  wmma_gemm_rowtile<<<1024, 256, 0, stream>>>(latlon, W1bt, nullptr, dstPre, 16384);

  // fused edge-MLP + aggregation + node-MLP
  fused_edge_node<<<2048, 256, SMEM_FUSED, stream>>>(
      srcPre, dstPre, W2t, W3t, W1nt, W2nt, W3nt, eb1, eb2, eg, ebe, nb0, nb1,
      nb2, ng, nbe, latlon, edge_src, edge_at, eW0, (float*)d_out);
}
